// FwdAttentionLayer_37288906064337
// MI455X (gfx1250) — matrix operand order repaired and checked by reading, not compiled, not run on hardware
//
#include <hip/hip_runtime.h>
#include <hip/hip_bf16.h>

typedef __attribute__((ext_vector_type(16))) _Float16 v16h;
typedef __attribute__((ext_vector_type(8)))  _Float16 v8h;
typedef __attribute__((ext_vector_type(8)))  float    v8f;

#define H_DIM 64
#define S_DIM 16
#define EF_DIM 8
#define IN_DIM 168          // 2H + 2S + EF
#define KPAD 192            // IN padded to multiple of 32
#define WID 128             // hidden width
#define A_STRIDE 200        // halfs per LDS row (400 B, 16-B aligned multiples)
#define WAVES 8

// ---------- helpers ----------
__device__ inline unsigned fenc(float f) {
    unsigned u = __float_as_uint(f);
    return (u & 0x80000000u) ? ~u : (u | 0x80000000u);
}
__device__ inline float fdec(unsigned u) {
    unsigned v = (u & 0x80000000u) ? (u & 0x7FFFFFFFu) : ~u;
    return __uint_as_float(v);
}
__device__ inline void st4h(_Float16* p, float4 v) {
    p[0] = (_Float16)v.x; p[1] = (_Float16)v.y;
    p[2] = (_Float16)v.z; p[3] = (_Float16)v.w;
}

// ---------- kernel 1: init outputs / segment buffers ----------
__global__ void k_init(float* agg, float* seg_sum, unsigned* seg_max_u,
                       int nagg, int nnodes) {
    int i = blockIdx.x * blockDim.x + threadIdx.x;
    if (i < nagg) agg[i] = 0.0f;
    if (i < nnodes) { seg_sum[i] = 0.0f; seg_max_u[i] = 0u; }
}

// ---------- kernel 2: W1 (f32 [128][168]) -> f16 padded [128][192] ----------
__global__ void k_prep(const float* __restrict__ W1, _Float16* __restrict__ W1h) {
    int i = blockIdx.x * blockDim.x + threadIdx.x;
    if (i >= WID * KPAD) return;
    int r = i / KPAD, c = i % KPAD;
    W1h[i] = (_Float16)(c < IN_DIM ? W1[r * IN_DIM + c] : 0.0f);
}

// ---------- kernel 3: WMMA edge-MLP -> scaled scores ----------
__global__ __launch_bounds__(256) void k_score(
    const float* __restrict__ h, const float* __restrict__ xs,
    const float* __restrict__ ef, const float* __restrict__ b1,
    const float* __restrict__ W2, const float* __restrict__ b2,
    const int* __restrict__ src, const int* __restrict__ dst,
    const _Float16* __restrict__ W1h, float* __restrict__ scaled, int E)
{
    __shared__ _Float16 As[WAVES * 16 * A_STRIDE];

    const int tid  = threadIdx.x;
    const int wave = tid >> 5;
    const int lane = tid & 31;
    const long tile = (long)blockIdx.x * WAVES + wave;

    _Float16* A = &As[wave * 16 * A_STRIDE];

    // ---- stage mlp_in for 16 edges into LDS (f32 -> f16) ----
    const int e_loc = lane & 15;
    long e_glob = tile * 16 + e_loc;
    if (e_glob >= E) e_glob = E - 1;           // clamp; padded scores not stored
    const int s = src[e_glob];
    const int d = dst[e_glob];
    _Float16* row = A + e_loc * A_STRIDE;

    if (lane < 16) {
        // cols 0..63 : h[src]
        const float4* hs4 = (const float4*)(h + (size_t)s * H_DIM);
        #pragma unroll
        for (int i = 0; i < 16; ++i) st4h(row + i * 4, hs4[i]);
        // cols 64..95 : h[dest][0..31]
        const float4* hd4 = (const float4*)(h + (size_t)d * H_DIM);
        #pragma unroll
        for (int i = 0; i < 8; ++i) st4h(row + 64 + i * 4, hd4[i]);
    } else {
        // cols 96..127 : h[dest][32..63]
        const float4* hd4 = (const float4*)(h + (size_t)d * H_DIM) + 8;
        #pragma unroll
        for (int i = 0; i < 8; ++i) st4h(row + 96 + i * 4, hd4[i]);
        // cols 128..143 : x_s[src]
        const float4* xss = (const float4*)(xs + (size_t)s * S_DIM);
        #pragma unroll
        for (int i = 0; i < 4; ++i) st4h(row + 128 + i * 4, xss[i]);
        // cols 144..159 : x_s[dest]
        const float4* xsd = (const float4*)(xs + (size_t)d * S_DIM);
        #pragma unroll
        for (int i = 0; i < 4; ++i) st4h(row + 144 + i * 4, xsd[i]);
        // cols 160..167 : edge_features
        const float4* efp = (const float4*)(ef + (size_t)e_glob * EF_DIM);
        #pragma unroll
        for (int i = 0; i < 2; ++i) st4h(row + 160 + i * 4, efp[i]);
        // cols 168..191 : zero pad
        #pragma unroll
        for (int i = IN_DIM; i < KPAD; ++i) row[i] = (_Float16)0.0f;
    }
    __syncthreads();

    // ---- GEMM: [16 x 192] x [192 x 128] via v_wmma_f32_16x16x32_f16 ----
    const int col  = lane & 15;                    // N position / A row
    const int koffA = (lane & 16) ? 8 : 0;         // A-fragment K split
    const int koffB = (lane & 16) ? 8 : 0;         // B-fragment K split
    const _Float16* Arow = A + col * A_STRIDE;

    float dotacc[8];
    #pragma unroll
    for (int r = 0; r < 8; ++r) dotacc[r] = 0.0f;

    #pragma unroll
    for (int nt = 0; nt < 8; ++nt) {
        v8f c = {};
        const _Float16* Wrow = W1h + (size_t)(nt * 16 + col) * KPAD;
        #pragma unroll
        for (int kt = 0; kt < 6; ++kt) {
            const int ka = kt * 32 + koffA;
            v8h a0 = *(const v8h*)(Arow + ka);
            v8h a1 = *(const v8h*)(Arow + ka + 16);
            const int kb = kt * 32 + koffB;
            v8h bv0 = *(const v8h*)(Wrow + kb);
            v8h bv1 = *(const v8h*)(Wrow + kb + 16);
            v16h av, bvv;
            #pragma unroll
            for (int i = 0; i < 8; ++i) {
                av[i] = a0[i];  av[i + 8] = a1[i];
                bvv[i] = bv0[i]; bvv[i + 8] = bv1[i];
            }
            c = __builtin_amdgcn_wmma_f32_16x16x32_f16(
                    false, av, false, bvv, (short)0, c, false, false);
        }
        // fused bias + relu + W2 dot on the f32 accumulators
        const int j = nt * 16 + col;
        const float bb = b1[j];
        const float w2 = W2[j];
        #pragma unroll
        for (int r = 0; r < 8; ++r) {
            float hid = c[r] + bb;
            hid = hid > 0.0f ? hid : 0.0f;
            dotacc[r] += hid * w2;
        }
    }

    // reduce across the 16 lanes of each half-wave (N dimension)
    #pragma unroll
    for (int r = 0; r < 8; ++r) {
        float v = dotacc[r];
        v += __shfl_xor(v, 1, 32);
        v += __shfl_xor(v, 2, 32);
        v += __shfl_xor(v, 4, 32);
        v += __shfl_xor(v, 8, 32);
        dotacc[r] = v;
    }

    // lane 0 -> edges 0..7, lane 16 -> edges 8..15
    if ((lane & 15) == 0) {
        const int mbase = (lane >> 4) * 8;
        const float b2v = b2[0];
        #pragma unroll
        for (int r = 0; r < 8; ++r) {
            long eg = tile * 16 + mbase + r;
            if (eg < E) {
                float sc = dotacc[r] + b2v;
                sc = sc > 0.0f ? sc : 0.01f * sc;   // leaky_relu(0.01)
                scaled[eg] = sc * 0.125f;           // / sqrt(64)
            }
        }
    }
}

// ---------- kernel 4: segment max ----------
__global__ void k_max(const float* __restrict__ scaled, const int* __restrict__ dst,
                      unsigned* __restrict__ seg_max_u, int E) {
    int e = blockIdx.x * blockDim.x + threadIdx.x;
    if (e >= E) return;
    atomicMax(&seg_max_u[dst[e]], fenc(scaled[e]));
}

// ---------- kernel 5: exp + segment sum ----------
__global__ void k_exp(const float* __restrict__ scaled, const int* __restrict__ dst,
                      const unsigned* __restrict__ seg_max_u,
                      float* __restrict__ exp_s, float* __restrict__ seg_sum, int E) {
    int e = blockIdx.x * blockDim.x + threadIdx.x;
    if (e >= E) return;
    float m = fdec(seg_max_u[dst[e]]);
    float ex = expf(scaled[e] - m);
    exp_s[e] = ex;
    atomicAdd(&seg_sum[dst[e]], ex);
}

// ---------- kernel 6: weights + scatter-add aggregation ----------
__global__ void k_final(const float* __restrict__ h, const int* __restrict__ src,
                        const int* __restrict__ dst,
                        const float* __restrict__ exp_s, const float* __restrict__ seg_sum,
                        float* __restrict__ agg, float* __restrict__ wout, long E) {
    long gid = (long)blockIdx.x * blockDim.x + threadIdx.x;
    long e = gid >> 6;           // 64 channels per edge
    int  k = (int)(gid & 63);
    if (e >= E) return;
    int d = dst[e];
    float w = exp_s[e] / (seg_sum[d] + 1e-9f);
    if (k == 0) wout[e] = w;
    atomicAdd(&agg[(size_t)d * H_DIM + k], h[(size_t)src[e] * H_DIM + k] * w);
}

// ---------- launcher ----------
extern "C" void kernel_launch(void* const* d_in, const int* in_sizes, int n_in,
                              void* d_out, int out_size, void* d_ws, size_t ws_size,
                              hipStream_t stream) {
    const float* h   = (const float*)d_in[0];
    const float* xs  = (const float*)d_in[1];
    const float* ef  = (const float*)d_in[2];
    const float* W1  = (const float*)d_in[3];
    const float* b1  = (const float*)d_in[4];
    const float* W2  = (const float*)d_in[5];
    const float* b2  = (const float*)d_in[6];
    const int*   src = (const int*)d_in[7];
    const int*   dst = (const int*)d_in[8];

    const int N = in_sizes[0] / H_DIM;       // 50000
    const int E = in_sizes[7];               // 800000

    // output layout: aggregated [N*64] then weights [E]
    float* agg  = (float*)d_out;
    float* wout = (float*)d_out + (size_t)N * H_DIM;

    // workspace layout
    char* w = (char*)d_ws;
    _Float16* W1h      = (_Float16*)w;                               // 49152 B
    float*    scaled   = (float*)(w + (size_t)WID * KPAD * 2);       // E*4
    float*    exp_s    = scaled + E;                                 // E*4
    float*    seg_sum  = exp_s + E;                                  // N*4
    unsigned* seg_max_u = (unsigned*)(seg_sum + N);                  // N*4

    const int nagg = N * H_DIM;
    k_init<<<(nagg + 255) / 256, 256, 0, stream>>>(agg, seg_sum, seg_max_u, nagg, N);
    k_prep<<<(WID * KPAD + 255) / 256, 256, 0, stream>>>(W1, W1h);

    const int ntiles  = (E + 15) / 16;
    const int nblocks = (ntiles + WAVES - 1) / WAVES;
    k_score<<<nblocks, 256, 0, stream>>>(h, xs, ef, b1, W2, b2, src, dst,
                                         W1h, scaled, E);

    k_max<<<(E + 255) / 256, 256, 0, stream>>>(scaled, dst, seg_max_u, E);
    k_exp<<<(E + 255) / 256, 256, 0, stream>>>(scaled, dst, seg_max_u, exp_s, seg_sum, E);

    const long tot = (long)E * H_DIM;
    k_final<<<(unsigned)((tot + 255) / 256), 256, 0, stream>>>(
        h, src, dst, exp_s, seg_sum, agg, wout, (long)E);
}
